// CausalTemporalAttention_86148454023465
// MI455X (gfx1250) — compile-verified, matrix operand order checked by reading, not run on hardware
//
#include <hip/hip_runtime.h>
#include <math.h>
#include <stdint.h>

typedef __attribute__((ext_vector_type(8)))  _Float16 v8h;
typedef __attribute__((ext_vector_type(16))) _Float16 v16h;
typedef __attribute__((ext_vector_type(8)))  float    v8f;
typedef __attribute__((ext_vector_type(4)))  unsigned int v4u;
typedef __attribute__((ext_vector_type(8)))  int      v8i;
typedef __attribute__((ext_vector_type(4)))  int      v4i;

#define DIMC   768
#define TSEQ   1024
#define BATCH  4
#define NHEADS 12
#define HD     64
#define MROWS  (BATCH * TSEQ)   // 4096
#define N3     (3 * DIMC)       // 2304
#define BH     (BATCH * NHEADS) // 48
#define NKC    (DIMC / 32)      // 24 k-chunks
#define LNEPS  1e-5f

// ---------------------------------------------------------------------------
// WMMA helpers (CDNA5 wave32, 16x16x32 f16 -> f32)
// ---------------------------------------------------------------------------
__device__ __forceinline__ v16h load_frag16(const _Float16* rowp, int hi8) {
  v8h a = *(const v8h*)(rowp + hi8);
  v8h b = *(const v8h*)(rowp + 16 + hi8);
  return __builtin_shufflevector(a, b, 0,1,2,3,4,5,6,7,8,9,10,11,12,13,14,15);
}
__device__ __forceinline__ v8f wmma_f16(v16h a, v16h b, v8f c) {
  return __builtin_amdgcn_wmma_f32_16x16x32_f16(false, a, false, b, (short)0, c,
                                                false, false);
}

// ---------------------------------------------------------------------------
// TDM: async 2D tile copy global->LDS (f16 elements).
// Tile: tile_rows rows x 32 halfs (64B) per row, global row stride
// row_stride_halfs. LDS rows padded to 40 halfs via TDM pad
// (pad_interval code 3 = every 16 DWORDs, pad_amount code 3 = 4 DWORDs).
// This toolchain's builtin takes 6 args: (v4u, v8i, v4i, v4i, v8i, i32 cpol).
// ---------------------------------------------------------------------------
__device__ __forceinline__ void tdm_load_tile(unsigned lds_byte_off,
                                              const _Float16* gptr,
                                              unsigned tile_rows,
                                              unsigned row_stride_halfs) {
  unsigned long long ga = (unsigned long long)(uintptr_t)gptr;
  v4u g0;
  g0[0] = 1u;                                   // count=1, user descriptor
  g0[1] = lds_byte_off;                         // lds_addr
  g0[2] = (unsigned)(ga & 0xFFFFFFFFu);         // global_addr lo
  g0[3] = (unsigned)((ga >> 32) & 0x01FFFFFFu) | (2u << 30);  // hi | type=2
  v8i g1;
  // data_size=2B (code1), pad_enable, pad_interval=3 (16 dw), pad_amount=3 (4 dw)
  g1[0] = (int)((1u << 16) | (1u << 20) | (3u << 22) | (3u << 25));
  unsigned td0 = row_stride_halfs;              // tensor_dim0 (>= tile_dim0)
  unsigned td1 = 1u << 20;                      // tensor_dim1 (never clip)
  g1[1] = (int)((td0 & 0xFFFFu) << 16);                      // abar=0 | td0.lo
  g1[2] = (int)(((td0 >> 16) & 0xFFFFu) | ((td1 & 0xFFFFu) << 16));
  g1[3] = (int)(((td1 >> 16) & 0xFFFFu) | (32u << 16));      // td1.hi | tile_dim0=32
  g1[4] = (int)(tile_rows & 0xFFFFu);                        // tile_dim1 | tile_dim2=0
  g1[5] = (int)row_stride_halfs;                             // dim0 stride lo32
  g1[6] = 0;                                                 // stride hi | dim1 stride
  g1[7] = 0;
  v4i g2 = {0, 0, 0, 0};
  v4i g3 = {0, 0, 0, 0};
  v8i g4 = {0, 0, 0, 0, 0, 0, 0, 0};
  __builtin_amdgcn_tensor_load_to_lds(g0, g1, g2, g3, g4, 0);
}
#define LDS_OFF(p) ((unsigned)(uintptr_t)(p))  // flat addr low 32 bits == LDS offset

// ---------------------------------------------------------------------------
// fp32 -> fp16 conversion (plain) and transposing conversion for weights
// ---------------------------------------------------------------------------
__global__ __launch_bounds__(256) void cvt_f32_f16(const float* __restrict__ s,
                                                   _Float16* __restrict__ d, int n) {
  int i = blockIdx.x * 256 + threadIdx.x;
  if (i < n) d[i] = (_Float16)s[i];
}
// s: [K][N] f32 -> d: [N][K] f16
__global__ __launch_bounds__(256) void cvt_transpose_f16(const float* __restrict__ s,
                                                         _Float16* __restrict__ d,
                                                         int K, int N) {
  int i = blockIdx.x * 256 + threadIdx.x;
  if (i < K * N) {
    int k = i / N, n = i - k * N;
    d[(size_t)n * K + k] = (_Float16)s[i];
  }
}

// ---------------------------------------------------------------------------
// QKV GEMM: [4096x768] x Wt[2304x768]^T + bias -> Q,K head-major; V d-major
// qkv layout: Q,K: [BH][TSEQ][HD]  V: [BH][HD][TSEQ]
// block 256 (8 waves), tile 128M x 128N, wave tile 64M x 32N, TDM double-buffer
// ---------------------------------------------------------------------------
__global__ __launch_bounds__(256)
void qkv_gemm_kernel(const _Float16* __restrict__ Xh,
                     const _Float16* __restrict__ Wt0, const _Float16* __restrict__ Wt1,
                     const float* __restrict__ b0, const float* __restrict__ b1,
                     _Float16* __restrict__ qkv0, _Float16* __restrict__ qkv1) {
  const int branch = blockIdx.z;
  const _Float16* Wt = branch ? Wt1 : Wt0;
  const float* bias = branch ? b1 : b0;
  _Float16* dst = branch ? qkv1 : qkv0;

  const int m0 = blockIdx.x * 128;
  const int n0 = blockIdx.y * 128;
  const int tid = threadIdx.x;
  const int wave = tid >> 5, lane = tid & 31;
  const int lane15 = lane & 15, hi8 = (lane & 16) ? 8 : 0;
  const int wm = wave >> 2, wn = wave & 3;

  __shared__ __align__(16) _Float16 As[2][128][40];
  __shared__ __align__(16) _Float16 Bs[2][128][40];

  v8f acc[4][2];
#pragma unroll
  for (int mt = 0; mt < 4; ++mt)
#pragma unroll
    for (int nt = 0; nt < 2; ++nt) acc[mt][nt] = (v8f){};

  if (wave == 0) {
    tdm_load_tile(LDS_OFF(&As[0][0][0]), Xh + (size_t)m0 * DIMC, 128, DIMC);
    tdm_load_tile(LDS_OFF(&Bs[0][0][0]), Wt + (size_t)n0 * DIMC, 128, DIMC);
    __builtin_amdgcn_s_wait_tensorcnt(0);
  }
  __syncthreads();

  for (int c = 0; c < NKC; ++c) {
    const int cur = c & 1, nxt = cur ^ 1;
    if ((c + 1 < NKC) && wave == 0) {
      tdm_load_tile(LDS_OFF(&As[nxt][0][0]), Xh + (size_t)m0 * DIMC + (c + 1) * 32, 128, DIMC);
      tdm_load_tile(LDS_OFF(&Bs[nxt][0][0]), Wt + (size_t)n0 * DIMC + (c + 1) * 32, 128, DIMC);
    }
    v16h af[4], bf[2];
#pragma unroll
    for (int mt = 0; mt < 4; ++mt)
      af[mt] = load_frag16(&As[cur][wm * 64 + mt * 16 + lane15][0], hi8);
#pragma unroll
    for (int nt = 0; nt < 2; ++nt)
      bf[nt] = load_frag16(&Bs[cur][wn * 32 + nt * 16 + lane15][0], hi8);
#pragma unroll
    for (int mt = 0; mt < 4; ++mt)
#pragma unroll
      for (int nt = 0; nt < 2; ++nt) acc[mt][nt] = wmma_f16(af[mt], bf[nt], acc[mt][nt]);
    if (wave == 0) __builtin_amdgcn_s_wait_tensorcnt(0);
    __syncthreads();
  }

  // epilogue: Q,K -> [three][BH][T][HD]; V -> [2][BH][HD][T]
#pragma unroll
  for (int mt = 0; mt < 4; ++mt) {
#pragma unroll
    for (int nt = 0; nt < 2; ++nt) {
#pragma unroll
      for (int r = 0; r < 8; ++r) {
        int m = m0 + wm * 64 + mt * 16 + r + hi8;
        int n = n0 + wn * 32 + nt * 16 + lane15;
        int three = n / DIMC, rem = n - three * DIMC;
        int h = rem >> 6, d = rem & 63;
        int b = m >> 10, t = m & 1023;
        size_t bhs = (size_t)(three * BH + b * NHEADS + h);
        size_t idx = (three == 2) ? (bhs * HD + d) * TSEQ + t
                                  : (bhs * TSEQ + t) * HD + d;
        dst[idx] = (_Float16)(acc[mt][nt][r] + bias[n]);
      }
    }
  }
}

// ---------------------------------------------------------------------------
// Flash attention: grid(16 qtiles, 48 b*h, 2 branch), block 128 (4 waves).
// Fully wave-independent (no block barriers): K and V fragments come straight
// from global (V is stored d-major), only P round-trips through wave-local LDS.
// ---------------------------------------------------------------------------
__global__ __launch_bounds__(128)
void attn_kernel(const _Float16* __restrict__ qkv0, const _Float16* __restrict__ qkv1,
                 _Float16* __restrict__ aout0, _Float16* __restrict__ aout1) {
  const int branch = blockIdx.z;  // 0 = causal, 1 = anti-causal
  const _Float16* qkv = branch ? qkv1 : qkv0;
  _Float16* aout = branch ? aout1 : aout0;

  const int bh = blockIdx.y;
  const int b = bh / NHEADS, h = bh - b * NHEADS;
  const int tid = threadIdx.x;
  const int wave = tid >> 5, lane = tid & 31;
  const int lane15 = lane & 15, hi8 = (lane & 16) ? 8 : 0;
  const int qrow0 = blockIdx.x * 64 + wave * 16;

  const size_t hdT = (size_t)TSEQ * HD;
  const _Float16* Qh = qkv + (size_t)bh * hdT;
  const _Float16* Kh = qkv + (size_t)(BH + bh) * hdT;        // [T][HD]
  const _Float16* Vh = qkv + (size_t)(2 * BH + bh) * hdT;    // [HD][T]

  __shared__ __align__(16) _Float16 Pl[4][16][40];

  v16h qa[2];
  {
    const _Float16* qrow = Qh + (size_t)(qrow0 + lane15) * HD;
#pragma unroll
    for (int dc = 0; dc < 2; ++dc) {
      v16h t = load_frag16(qrow + dc * 32, hi8);
#pragma unroll
      for (int i = 0; i < 16; ++i) qa[dc][i] = (_Float16)((float)t[i] * 0.125f);
    }
  }

  v8f o[4];
#pragma unroll
  for (int dt = 0; dt < 4; ++dt) o[dt] = (v8f){};
  float mrow[8], lrow[8];
#pragma unroll
  for (int r = 0; r < 8; ++r) { mrow[r] = -1e30f; lrow[r] = 0.f; }

  const int kc0 = branch ? (qrow0 & ~31) : 0;
  const int kc1 = branch ? (TSEQ - 32) : (qrow0 & ~31);

  for (int kc = kc0; kc <= kc1; kc += 32) {
    // scores: Q(16x64) @ K^T -> 16 x 32
    v8f s0 = (v8f){}, s1 = (v8f){};
#pragma unroll
    for (int dc = 0; dc < 2; ++dc) {
      v16h k0 = load_frag16(Kh + (size_t)(kc + lane15) * HD + dc * 32, hi8);
      v16h k1 = load_frag16(Kh + (size_t)(kc + 16 + lane15) * HD + dc * 32, hi8);
      s0 = wmma_f16(qa[dc], k0, s0);
      s1 = wmma_f16(qa[dc], k1, s1);
    }

    // mask + online softmax
#pragma unroll
    for (int r = 0; r < 8; ++r) {
      int i = qrow0 + r + hi8;
      int j0 = kc + lane15, j1 = kc + 16 + lane15;
      float a = s0[r], c = s1[r];
      if (branch ? (j0 < i) : (j0 > i)) a = -1e30f;
      if (branch ? (j1 < i) : (j1 > i)) c = -1e30f;
      float mx = fmaxf(a, c);
#pragma unroll
      for (int off = 8; off; off >>= 1) mx = fmaxf(mx, __shfl_xor(mx, off, 32));
      float mnew = fmaxf(mrow[r], mx);
      float p0 = (a > -5e29f) ? __expf(a - mnew) : 0.f;
      float p1 = (c > -5e29f) ? __expf(c - mnew) : 0.f;
      float corr = __expf(mrow[r] - mnew);
      float rs = p0 + p1;
#pragma unroll
      for (int off = 8; off; off >>= 1) rs += __shfl_xor(rs, off, 32);
      lrow[r] = lrow[r] * corr + rs;
      mrow[r] = mnew;
#pragma unroll
      for (int dt = 0; dt < 4; ++dt) o[dt][r] *= corr;
      Pl[wave][r + hi8][lane15] = (_Float16)p0;
      Pl[wave][r + hi8][16 + lane15] = (_Float16)p1;
    }
    asm volatile("s_wait_dscnt 0" ::: "memory");  // wave-local P store->load

    // O += P(16x32) @ V(32x64), V fragments direct from global (d-major)
    v16h pa = load_frag16(&Pl[wave][lane15][0], hi8);
#pragma unroll
    for (int dt = 0; dt < 4; ++dt) {
      v16h vb = load_frag16(Vh + (size_t)(dt * 16 + lane15) * TSEQ + kc, hi8);
      o[dt] = wmma_f16(pa, vb, o[dt]);
    }
  }

  // normalize + store to [B*T, 768] f16 (col = h*64 + d)
#pragma unroll
  for (int r = 0; r < 8; ++r) {
    float inv = 1.f / lrow[r];
    int row = b * TSEQ + qrow0 + r + hi8;
#pragma unroll
    for (int dt = 0; dt < 4; ++dt)
      aout[(size_t)row * DIMC + h * HD + dt * 16 + lane15] = (_Float16)(o[dt][r] * inv);
  }
}

// ---------------------------------------------------------------------------
// Proj GEMM (both branches fused, 48 k-chunks) + bias + residual -> y (fp32)
// ---------------------------------------------------------------------------
__global__ __launch_bounds__(256)
void proj_gemm_kernel(const _Float16* __restrict__ A0, const _Float16* __restrict__ A1,
                      const _Float16* __restrict__ Wt0, const _Float16* __restrict__ Wt1,
                      const float* __restrict__ bp0, const float* __restrict__ bp1,
                      const float* __restrict__ X, float* __restrict__ Y) {
  const int m0 = blockIdx.x * 128;
  const int n0 = blockIdx.y * 128;
  const int tid = threadIdx.x;
  const int wave = tid >> 5, lane = tid & 31;
  const int lane15 = lane & 15, hi8 = (lane & 16) ? 8 : 0;
  const int wm = wave >> 2, wn = wave & 3;

  __shared__ __align__(16) _Float16 As[2][128][40];
  __shared__ __align__(16) _Float16 Bs[2][128][40];

  v8f acc[4][2];
#pragma unroll
  for (int mt = 0; mt < 4; ++mt)
#pragma unroll
    for (int nt = 0; nt < 2; ++nt) acc[mt][nt] = (v8f){};

  const int NC = 2 * NKC;  // 48 chunks: branch = c / NKC
  auto srcA = [&](int c) {
    const _Float16* A = (c < NKC) ? A0 : A1;
    return A + (size_t)m0 * DIMC + (c % NKC) * 32;
  };
  auto srcB = [&](int c) {
    const _Float16* W = (c < NKC) ? Wt0 : Wt1;
    return W + (size_t)n0 * DIMC + (c % NKC) * 32;
  };

  if (wave == 0) {
    tdm_load_tile(LDS_OFF(&As[0][0][0]), srcA(0), 128, DIMC);
    tdm_load_tile(LDS_OFF(&Bs[0][0][0]), srcB(0), 128, DIMC);
    __builtin_amdgcn_s_wait_tensorcnt(0);
  }
  __syncthreads();

  for (int c = 0; c < NC; ++c) {
    const int cur = c & 1, nxt = cur ^ 1;
    if ((c + 1 < NC) && wave == 0) {
      tdm_load_tile(LDS_OFF(&As[nxt][0][0]), srcA(c + 1), 128, DIMC);
      tdm_load_tile(LDS_OFF(&Bs[nxt][0][0]), srcB(c + 1), 128, DIMC);
    }
    v16h af[4], bf[2];
#pragma unroll
    for (int mt = 0; mt < 4; ++mt)
      af[mt] = load_frag16(&As[cur][wm * 64 + mt * 16 + lane15][0], hi8);
#pragma unroll
    for (int nt = 0; nt < 2; ++nt)
      bf[nt] = load_frag16(&Bs[cur][wn * 32 + nt * 16 + lane15][0], hi8);
#pragma unroll
    for (int mt = 0; mt < 4; ++mt)
#pragma unroll
      for (int nt = 0; nt < 2; ++nt) acc[mt][nt] = wmma_f16(af[mt], bf[nt], acc[mt][nt]);
    if (wave == 0) __builtin_amdgcn_s_wait_tensorcnt(0);
    __syncthreads();
  }

#pragma unroll
  for (int mt = 0; mt < 4; ++mt) {
#pragma unroll
    for (int nt = 0; nt < 2; ++nt) {
#pragma unroll
      for (int r = 0; r < 8; ++r) {
        int m = m0 + wm * 64 + mt * 16 + r + hi8;
        int n = n0 + wn * 32 + nt * 16 + lane15;
        size_t idx = (size_t)m * DIMC + n;
        Y[idx] = acc[mt][nt][r] + bp0[n] + bp1[n] + X[idx];
      }
    }
  }
}

// ---------------------------------------------------------------------------
// LayerNorm over last dim (768), one block per row
// ---------------------------------------------------------------------------
__global__ __launch_bounds__(256)
void ln_kernel(const float* __restrict__ Y, const float* __restrict__ g,
               const float* __restrict__ be, float* __restrict__ out) {
  const int row = blockIdx.x;
  const float* yr = Y + (size_t)row * DIMC;
  const int tid = threadIdx.x, wave = tid >> 5, lane = tid & 31;
  float s = 0.f, ss = 0.f;
  for (int i = tid; i < DIMC; i += 256) { float v = yr[i]; s += v; ss += v * v; }
#pragma unroll
  for (int off = 16; off; off >>= 1) { s += __shfl_xor(s, off, 32); ss += __shfl_xor(ss, off, 32); }
  __shared__ float red[16];
  if (lane == 0) { red[wave] = s; red[8 + wave] = ss; }
  __syncthreads();
  float ts = 0.f, tss = 0.f;
#pragma unroll
  for (int i = 0; i < 8; ++i) { ts += red[i]; tss += red[8 + i]; }
  float mu = ts * (1.f / DIMC);
  float var = tss * (1.f / DIMC) - mu * mu;
  float rstd = rsqrtf(var + LNEPS);
  for (int i = tid; i < DIMC; i += 256)
    out[(size_t)row * DIMC + i] = (yr[i] - mu) * rstd * g[i] + be[i];
}

// ---------------------------------------------------------------------------
extern "C" void kernel_launch(void* const* d_in, const int* in_sizes, int n_in,
                              void* d_out, int out_size, void* d_ws, size_t ws_size,
                              hipStream_t stream) {
  const float* x       = (const float*)d_in[0];
  const float* Wqkv_c  = (const float*)d_in[1];
  const float* bqkv_c  = (const float*)d_in[2];
  const float* Wp_c    = (const float*)d_in[3];
  const float* bp_c    = (const float*)d_in[4];
  const float* Wqkv_ac = (const float*)d_in[5];
  const float* bqkv_ac = (const float*)d_in[6];
  const float* Wp_ac   = (const float*)d_in[7];
  const float* bp_ac   = (const float*)d_in[8];
  const float* gamma   = (const float*)d_in[9];
  const float* beta    = (const float*)d_in[10];
  float* out = (float*)d_out;

  const size_t NX    = (size_t)MROWS * DIMC;
  const size_t NWQKV = (size_t)DIMC * N3;
  const size_t NWP   = (size_t)DIMC * DIMC;
  const size_t NQKV  = (size_t)3 * BH * TSEQ * HD;

  char* p = (char*)d_ws;
  auto take = [&](size_t bytes) { char* r = p; p += (bytes + 255) & ~(size_t)255; return r; };
  _Float16* xh     = (_Float16*)take(NX * 2);
  _Float16* wqkv0t = (_Float16*)take(NWQKV * 2);  // [2304][768]
  _Float16* wqkv1t = (_Float16*)take(NWQKV * 2);
  _Float16* wp0t   = (_Float16*)take(NWP * 2);    // [768][768]
  _Float16* wp1t   = (_Float16*)take(NWP * 2);
  _Float16* qkv0   = (_Float16*)take(NQKV * 2);
  _Float16* qkv1   = (_Float16*)take(NQKV * 2);
  _Float16* aout0  = (_Float16*)take(NX * 2);
  _Float16* aout1  = (_Float16*)take(NX * 2);
  float*    ybuf   = (float*)take(NX * 4);

  cvt_f32_f16<<<(unsigned)((NX + 255) / 256), 256, 0, stream>>>(x, xh, (int)NX);
  cvt_transpose_f16<<<(unsigned)((NWQKV + 255) / 256), 256, 0, stream>>>(Wqkv_c, wqkv0t, DIMC, N3);
  cvt_transpose_f16<<<(unsigned)((NWQKV + 255) / 256), 256, 0, stream>>>(Wqkv_ac, wqkv1t, DIMC, N3);
  cvt_transpose_f16<<<(unsigned)((NWP + 255) / 256), 256, 0, stream>>>(Wp_c, wp0t, DIMC, DIMC);
  cvt_transpose_f16<<<(unsigned)((NWP + 255) / 256), 256, 0, stream>>>(Wp_ac, wp1t, DIMC, DIMC);

  qkv_gemm_kernel<<<dim3(MROWS / 128, N3 / 128, 2), 256, 0, stream>>>(
      xh, wqkv0t, wqkv1t, bqkv_c, bqkv_ac, qkv0, qkv1);

  attn_kernel<<<dim3(TSEQ / 64, BH, 2), 128, 0, stream>>>(qkv0, qkv1, aout0, aout1);

  proj_gemm_kernel<<<dim3(MROWS / 128, DIMC / 128), 256, 0, stream>>>(
      aout0, aout1, wp0t, wp1t, bp_c, bp_ac, x, ybuf);

  ln_kernel<<<MROWS, 256, 0, stream>>>(ybuf, gamma, beta, out);
}
